// Transformer_70557722739192
// MI455X (gfx1250) — matrix operand order repaired and checked by reading, not compiled, not run on hardware
//
#include <hip/hip_runtime.h>
#include <math.h>

typedef __bf16 bf16_t;
typedef __attribute__((ext_vector_type(16))) __bf16 v16bf;
typedef __attribute__((ext_vector_type(8)))  float  v8f;
typedef __attribute__((ext_vector_type(4)))  unsigned int v4u;
typedef __attribute__((ext_vector_type(8)))  int v8i;
typedef __attribute__((ext_vector_type(4)))  int v4i;

#define EMB   1024
#define HEADS 16
#define HDIM  64
#define SEQ   2048
#define BATCH 2
#define TOK   (BATCH * SEQ)   // 4096 token rows
#define FF    (4 * EMB)       // 4096
#define VOCAB 32000

enum { GF_GELU = 1, GF_RES = 2, GF_OUTF = 4, GF_OUTB = 8, GF_QKV = 16 };

static __device__ __forceinline__ v8f wmma_bf16(v16bf a, v16bf b, v8f c) {
  // D = A(16x32) * B(32x16) + C(16x16), f32 accumulate
  return __builtin_amdgcn_wmma_f32_16x16x32_bf16(false, a, false, b, (short)0, c,
                                                 false, false);
}

// ---- WMMA fragment loaders (contiguous-k order, same for A and B, wave32) -------------------
// A (16x32 bf16): lane<16 row M=lane, elems 0..15 = K k0..k0+15
//                 lane>=16 row M=lane-16, elems 0..15 = k0+16..31
static __device__ __forceinline__ v16bf load_a_frag(const bf16_t* __restrict__ A,
                                                    int ld, int row0, int k0) {
  const int lane = threadIdx.x & 31;
  const bf16_t* p = A + (size_t)(row0 + (lane & 15)) * ld + k0 + ((lane & 16) ? 16 : 0);
  v16bf a;
#pragma unroll
  for (int i = 0; i < 8; ++i) { a[i] = p[i]; a[i + 8] = p[i + 8]; }
  return a;
}

// B (32x16 bf16) from a [N][K] operand:
// lane<16 col N=lane, elems = K k0..k0+15 ; lane>=16 col N=lane-16, elems = k0+16..31
static __device__ __forceinline__ v16bf load_b_frag(const bf16_t* __restrict__ Bt,
                                                    int ld, int col0, int k0) {
  const int lane = threadIdx.x & 31;
  const bf16_t* p = Bt + (size_t)(col0 + (lane & 15)) * ld + k0 + ((lane & 16) ? 16 : 0);
  v16bf b;
#pragma unroll
  for (int i = 0; i < 16; ++i) b[i] = p[i];
  return b;
}

// A-frag of V^T: row m = d (within 16-chunk), K dim = 32 keys starting at k0.
static __device__ __forceinline__ v16bf load_vt_frag(const bf16_t* __restrict__ V,
                                                     int k0, int d0) {
  const int lane = threadIdx.x & 31;
  const int d  = d0 + (lane & 15);
  const int kb = k0 + ((lane & 16) ? 16 : 0);
  v16bf a;
#pragma unroll
  for (int i = 0; i < 8; ++i) {
    a[i]     = V[(size_t)(kb + i) * HDIM + d];
    a[i + 8] = V[(size_t)(kb + 8 + i) * HDIM + d];
  }
  return a;
}

// ---- Tensor Data Mover: stage a 2D bf16 tile (tile_d0 x tile_d1) into LDS --
// Row pitch = stride0 elements; D# per ISA 8.3/8.4 (2D: groups 2/3 zero).
static __device__ __forceinline__ void tdm_load_2d(
    unsigned lds_byte_off, const void* gptr,
    unsigned tensor_d0, unsigned tensor_d1,
    unsigned tile_d0, unsigned tile_d1, unsigned stride0_elems) {
  const unsigned long long ga = (unsigned long long)gptr;
  v4u g0;
  g0.x = 1u;                                   // count=1 valid, user descriptor
  g0.y = lds_byte_off;                         // lds_addr [63:32]
  g0.z = (unsigned)ga;                         // global_addr[31:0]
  g0.w = (unsigned)((ga >> 32) & 0x01FFFFFFu)  // global_addr[56:32]
         | (2u << 30);                         // type=2 ("image")
  v8i g1;
  g1[0] = (int)(1u << 16);                     // data_size=1 -> 2 bytes/elem
  g1[1] = (int)((tensor_d0 & 0xFFFFu) << 16);  // tensor_dim0[15:0] @ [63:48]
  g1[2] = (int)((tensor_d0 >> 16) | ((tensor_d1 & 0xFFFFu) << 16));
  g1[3] = (int)((tensor_d1 >> 16) | (tile_d0 << 16));   // tile_dim0 @ [127:112]
  g1[4] = (int)tile_d1;                        // tile_dim1 ; tile_dim2=0 (2D)
  g1[5] = (int)stride0_elems;                  // tensor_dim0_stride[31:0]
  g1[6] = 0;
  g1[7] = 0;
  const v4i z4 = {0, 0, 0, 0};
#if defined(__clang_major__) && (__clang_major__ >= 23)
  const v8i z8 = {0, 0, 0, 0, 0, 0, 0, 0};
  __builtin_amdgcn_tensor_load_to_lds(g0, g1, z4, z4, z8, 0);
#else
  __builtin_amdgcn_tensor_load_to_lds(g0, g1, z4, z4, 0);
#endif
}

// ---- fp32 -> bf16 transposing weight convert: W[K][N] -> Wt[N][K] ---------
__global__ void __launch_bounds__(256) transpose_f32_to_bf16(
    const float* __restrict__ W, bf16_t* __restrict__ Wt, int K, int N) {
  __shared__ float tile[32][33];
  const int kb = blockIdx.x * 32, nb = blockIdx.y * 32;
  const int tx = threadIdx.x & 31, ty = threadIdx.x >> 5;
#pragma unroll
  for (int r = 0; r < 4; ++r)
    tile[ty + 8 * r][tx] = W[(size_t)(kb + ty + 8 * r) * N + nb + tx];
  __syncthreads();
#pragma unroll
  for (int r = 0; r < 4; ++r)
    Wt[(size_t)(nb + ty + 8 * r) * K + kb + tx] = (bf16_t)tile[tx][ty + 8 * r];
}

// ---- rowwise LayerNorm: fp32 in -> bf16 out -------------------------------
__global__ void __launch_bounds__(256) layernorm_bf16_kernel(
    const float* __restrict__ in, const float* __restrict__ w,
    const float* __restrict__ b, bf16_t* __restrict__ out) {
  __shared__ float s1[256], s2[256];
  const int row = blockIdx.x;
  const float* x = in + (size_t)row * EMB;
  float a = 0.f, a2 = 0.f;
  for (int i = threadIdx.x; i < EMB; i += 256) {
    float v = x[i];
    a += v; a2 += v * v;
  }
  s1[threadIdx.x] = a; s2[threadIdx.x] = a2;
  __syncthreads();
  for (int s = 128; s > 0; s >>= 1) {
    if ((int)threadIdx.x < s) {
      s1[threadIdx.x] += s1[threadIdx.x + s];
      s2[threadIdx.x] += s2[threadIdx.x + s];
    }
    __syncthreads();
  }
  const float mean = s1[0] * (1.0f / EMB);
  const float var  = s2[0] * (1.0f / EMB) - mean * mean;
  const float rstd = rsqrtf(var + 1e-5f);
  bf16_t* o = out + (size_t)row * EMB;
  for (int i = threadIdx.x; i < EMB; i += 256)
    o[i] = (bf16_t)(w[i] * ((x[i] - mean) * rstd) + b[i]);
}

// ---- TDM-staged bf16 WMMA GEMM:  C[M][N] = A[M][K] * Bt[N][K]^T -----------
// 256 threads = 8 waves; macro tile 128x128; per wave 64(M)x32(N);
// K staged in 64-wide chunks by the Tensor Data Mover, double-buffered LDS.
// LDS: buf(0/1) x { A 128x64 bf16 (16KB) , B 128x64 bf16 (16KB) } = 64KB.
__global__ void __launch_bounds__(256) gemm_bf16_kernel(
    const bf16_t* __restrict__ A, const bf16_t* __restrict__ Bt,
    const float* __restrict__ bias, const float* __restrict__ residual,
    float* __restrict__ outF, bf16_t* __restrict__ outB,
    int M, int N, int K, int flags) {
  extern __shared__ char smem[];
  const int wave = threadIdx.x >> 5;
  const int lane = threadIdx.x & 31;
  const int wm = wave & 1;        // 2 wave rows  (64 M each)
  const int wn = wave >> 1;       // 4 wave cols  (32 N each)
  const int m_blk = blockIdx.x * 128;
  const int n_blk = blockIdx.y * 128;
  const int nk = K >> 6;          // number of 64-wide K chunks

  auto issue = [&](int kc, int buf) {
    const unsigned base = (unsigned)(uintptr_t)(smem) + (unsigned)(buf * 32768);
    tdm_load_2d(base, A + (size_t)m_blk * K + (size_t)kc * 64,
                (unsigned)K, (unsigned)M, 64u, 128u, (unsigned)K);
    tdm_load_2d(base + 16384u, Bt + (size_t)n_blk * K + (size_t)kc * 64,
                (unsigned)K, (unsigned)N, 64u, 128u, (unsigned)K);
  };

  if (wave == 0) issue(0, 0);

  const v8f z8 = {0.f, 0.f, 0.f, 0.f, 0.f, 0.f, 0.f, 0.f};
  v8f acc[4][2] = {{z8, z8}, {z8, z8}, {z8, z8}, {z8, z8}};

  for (int i = 0; i < nk; ++i) {
    const int buf = i & 1;
    if (wave == 0) {
      if (i + 1 < nk) {
        issue(i + 1, buf ^ 1);                      // prefetch next chunk
        __builtin_amdgcn_s_wait_tensorcnt((short)2); // current chunk landed
      } else {
        __builtin_amdgcn_s_wait_tensorcnt((short)0);
      }
    }
    __syncthreads();   // LDS tiles valid for everyone

    const bf16_t* lA = (const bf16_t*)(smem + buf * 32768);
    const bf16_t* lB = (const bf16_t*)(smem + buf * 32768 + 16384);
#pragma unroll
    for (int ks = 0; ks < 64; ks += 32) {
      v16bf af[4], bfr[2];
#pragma unroll
      for (int mi = 0; mi < 4; ++mi)
        af[mi] = load_a_frag(lA, 64, wm * 64 + mi * 16, ks);
#pragma unroll
      for (int nj = 0; nj < 2; ++nj)
        bfr[nj] = load_b_frag(lB, 64, wn * 32 + nj * 16, ks);
#pragma unroll
      for (int mi = 0; mi < 4; ++mi)
#pragma unroll
        for (int nj = 0; nj < 2; ++nj)
          acc[mi][nj] = wmma_bf16(af[mi], bfr[nj], acc[mi][nj]);
    }
    __syncthreads();   // done reading buf before it is overwritten
  }

#pragma unroll
  for (int mi = 0; mi < 4; ++mi) {
#pragma unroll
    for (int nj = 0; nj < 2; ++nj) {
      const int n  = n_blk + wn * 32 + nj * 16 + (lane & 15);
      const int mb = m_blk + wm * 64 + mi * 16 + ((lane & 16) ? 8 : 0);
      const float bv = bias ? bias[n] : 0.0f;
#pragma unroll
      for (int t = 0; t < 8; ++t) {
        const int m = mb + t;
        float val = acc[mi][nj][t] + bv;
        if (flags & GF_GELU) {
          const float u = val;
          val = 0.5f * u *
                (1.0f + tanhf(0.7978845608028654f * (u + 0.044715f * u * u * u)));
        }
        if (flags & GF_RES)  val += residual[(size_t)m * N + n];
        if (flags & GF_OUTF) outF[(size_t)m * N + n] = val;
        if (flags & GF_OUTB) outB[(size_t)m * N + n] = (bf16_t)val;
        if (flags & GF_QKV) {
          const int which = n >> 10, e = n & 1023;
          const int head = e >> 6, d = e & 63;
          const int bb = m >> 11, s = m & 2047;
          outB[(size_t)which * ((size_t)TOK * EMB) +
               (((size_t)(bb * HEADS + head) * SEQ + s) * HDIM) + d] = (bf16_t)val;
        }
      }
    }
  }
}

// ---- causal flash attention, one wave per 16-query tile per (b,h) ---------
// q,k,v: bf16 [B][H][S][64].  Writes h = x + ctx (fp32, [B][S][EMB]).
__global__ void __launch_bounds__(32) flash_attn_kernel(
    const bf16_t* __restrict__ q, const bf16_t* __restrict__ k,
    const bf16_t* __restrict__ v, const float* __restrict__ x,
    float* __restrict__ h) {
  const int lane  = threadIdx.x & 31;
  const int half  = (lane >> 4) & 1;
  const int qt    = blockIdx.x * 16;
  const int head  = blockIdx.y;
  const int batch = blockIdx.z;
  const size_t hb = (size_t)(batch * HEADS + head) * SEQ * HDIM;
  const bf16_t* Q  = q + hb;
  const bf16_t* Kp = k + hb;
  const bf16_t* Vp = v + hb;

  // Q as B-fragments (Q^T columns = queries), pre-scaled by 1/sqrt(64)
  const int qrow = qt + (lane & 15);
  v16bf bq0, bq1;
#pragma unroll
  for (int i = 0; i < 16; ++i) {
    bq0[i] = (bf16_t)((float)Q[(size_t)qrow * HDIM + (half ? 16 : 0) + i] * 0.125f);
    bq1[i] = (bf16_t)((float)Q[(size_t)qrow * HDIM + 32 + (half ? 16 : 0) + i] * 0.125f);
  }

  const v8f z8 = {0.f, 0.f, 0.f, 0.f, 0.f, 0.f, 0.f, 0.f};
  v8f o[4] = {z8, z8, z8, z8};   // O^T chunks: d 0-15,16-31,32-47,48-63
  float m_run = -3.0e38f, l_run = 0.0f;

  for (int k0 = 0; k0 <= qt + 15; k0 += 32) {
    if (k0 + 32 <= qt + 15) {    // hint next K/V tiles toward the caches
      __builtin_prefetch(Kp + (size_t)(k0 + 32) * HDIM, 0, 1);
      __builtin_prefetch(Vp + (size_t)(k0 + 32) * HDIM, 0, 1);
    }
    // S^T(32 keys x 16 queries) = K_tile * Q^T, K-dim = d (2 steps of 32)
    v16bf ka0  = load_a_frag(Kp, HDIM, k0,      0);
    v16bf ka0b = load_a_frag(Kp, HDIM, k0,      32);
    v16bf ka1  = load_a_frag(Kp, HDIM, k0 + 16, 0);
    v16bf ka1b = load_a_frag(Kp, HDIM, k0 + 16, 32);
    v8f s0 = wmma_bf16(ka0b, bq1, wmma_bf16(ka0, bq0, z8)); // keys k0..k0+15
    v8f s1 = wmma_bf16(ka1b, bq1, wmma_bf16(ka1, bq0, z8)); // keys k0+16..k0+31

    const int qi  = qt + (lane & 15);
    const int kb0 = k0 + (half ? 8 : 0);
    const int kb1 = k0 + 16 + (half ? 8 : 0);
    if (k0 + 31 > qt) {   // causal mask in diagonal region
#pragma unroll
      for (int t = 0; t < 8; ++t) {
        if (kb0 + t > qi) s0[t] = -3.0e38f;
        if (kb1 + t > qi) s1[t] = -3.0e38f;
      }
    }

    // online softmax; lanes (qcol) and (qcol+16) hold the same query
    float mx = m_run;
#pragma unroll
    for (int t = 0; t < 8; ++t) { mx = fmaxf(mx, s0[t]); mx = fmaxf(mx, s1[t]); }
    mx = fmaxf(mx, __shfl_xor(mx, 16));
    const float scale_old = __expf(m_run - mx);
    float p0[8], p1[8];
    float lsum = 0.0f;
#pragma unroll
    for (int t = 0; t < 8; ++t) {
      p0[t] = __expf(s0[t] - mx);
      p1[t] = __expf(s1[t] - mx);
      lsum += p0[t] + p1[t];
    }
    lsum += __shfl_xor(lsum, 16);
    l_run = l_run * scale_old + lsum;
    m_run = mx;
#pragma unroll
    for (int c = 0; c < 4; ++c)
#pragma unroll
      for (int t = 0; t < 8; ++t) o[c][t] *= scale_old;

    // Rebuild P^T as a B-fragment (cols = queries, K = 32 keys) via xor-16
    v16bf pb;
#pragma unroll
    for (int t = 0; t < 8; ++t) {
      const float q0 = __shfl_xor(p0[t], 16);
      const float q1 = __shfl_xor(p1[t], 16);
      if (half == 0) { pb[t] = (bf16_t)p0[t]; pb[t + 8] = (bf16_t)q0; }
      else           { pb[t] = (bf16_t)q1;    pb[t + 8] = (bf16_t)p1[t]; }
    }

    // O^T += V^T(d x 32keys) * P^T
    v16bf vt0 = load_vt_frag(Vp, k0, 0);
    v16bf vt1 = load_vt_frag(Vp, k0, 16);
    v16bf vt2 = load_vt_frag(Vp, k0, 32);
    v16bf vt3 = load_vt_frag(Vp, k0, 48);
    o[0] = wmma_bf16(vt0, pb, o[0]);
    o[1] = wmma_bf16(vt1, pb, o[1]);
    o[2] = wmma_bf16(vt2, pb, o[2]);
    o[3] = wmma_bf16(vt3, pb, o[3]);
  }

  // h = x + ctx ; O^T layout: lane = query col, vgpr = d row (+8 upper half)
  const float inv_l = 1.0f / l_run;
  const size_t rowbase = ((size_t)batch * SEQ + (size_t)qrow) * EMB + head * HDIM;
#pragma unroll
  for (int c = 0; c < 4; ++c) {
#pragma unroll
    for (int t = 0; t < 8; ++t) {
      const int d = c * 16 + (half ? 8 : 0) + t;
      h[rowbase + d] = x[rowbase + d] + o[c][t] * inv_l;
    }
  }
}

// ---------------------------------------------------------------------------
extern "C" void kernel_launch(void* const* d_in, const int* in_sizes, int n_in,
                              void* d_out, int out_size, void* d_ws, size_t ws_size,
                              hipStream_t stream) {
  (void)in_sizes; (void)n_in; (void)out_size; (void)ws_size;
  const float* x     = (const float*)d_in[0];
  const float* W_q   = (const float*)d_in[1];
  const float* W_k   = (const float*)d_in[2];
  const float* W_v   = (const float*)d_in[3];
  const float* ln1w  = (const float*)d_in[4];
  const float* ln1b  = (const float*)d_in[5];
  const float* ln2w  = (const float*)d_in[6];
  const float* ln2b  = (const float*)d_in[7];
  const float* ffw1  = (const float*)d_in[8];
  const float* ffb1  = (const float*)d_in[9];
  const float* ffw2  = (const float*)d_in[10];
  const float* ffb2  = (const float*)d_in[11];
  const float* projw = (const float*)d_in[12];
  const float* projb = (const float*)d_in[13];
  float* out = (float*)d_out;

  char* ws = (char*)d_ws;
  size_t off = 0;
  auto alloc = [&](size_t bytes) -> char* {
    char* p = ws + off;
    off += (bytes + 255) & ~(size_t)255;
    return p;
  };
  bf16_t* ln1o  = (bf16_t*)alloc((size_t)TOK * EMB * 2);
  bf16_t* WqkvT = (bf16_t*)alloc((size_t)3 * EMB * EMB * 2);
  bf16_t* qkv   = (bf16_t*)alloc((size_t)3 * TOK * EMB * 2);
  float*  hbuf  = (float*) alloc((size_t)TOK * EMB * 4);
  bf16_t* ln2o  = (bf16_t*)alloc((size_t)TOK * EMB * 2);
  bf16_t* w1T   = (bf16_t*)alloc((size_t)EMB * FF * 2);
  bf16_t* f1    = (bf16_t*)alloc((size_t)TOK * FF * 2);
  bf16_t* w2T   = (bf16_t*)alloc((size_t)FF * EMB * 2);
  bf16_t* h2b   = (bf16_t*)alloc((size_t)TOK * EMB * 2);
  bf16_t* projT = (bf16_t*)alloc((size_t)EMB * VOCAB * 2);

  const size_t LDS_BYTES = 65536;  // 2 x (A 16KB + B 16KB) double buffer

  // 1) weight convert + transpose to bf16 [N][K]
  transpose_f32_to_bf16<<<dim3(EMB / 32, EMB / 32), 256, 0, stream>>>(
      W_q, WqkvT, EMB, EMB);
  transpose_f32_to_bf16<<<dim3(EMB / 32, EMB / 32), 256, 0, stream>>>(
      W_k, WqkvT + (size_t)EMB * EMB, EMB, EMB);
  transpose_f32_to_bf16<<<dim3(EMB / 32, EMB / 32), 256, 0, stream>>>(
      W_v, WqkvT + (size_t)2 * EMB * EMB, EMB, EMB);
  transpose_f32_to_bf16<<<dim3(EMB / 32, FF / 32), 256, 0, stream>>>(
      ffw1, w1T, EMB, FF);
  transpose_f32_to_bf16<<<dim3(FF / 32, EMB / 32), 256, 0, stream>>>(
      ffw2, w2T, FF, EMB);
  transpose_f32_to_bf16<<<dim3(EMB / 32, VOCAB / 32), 256, 0, stream>>>(
      projw, projT, EMB, VOCAB);

  // 2) ln1(x) -> bf16
  layernorm_bf16_kernel<<<TOK, 256, 0, stream>>>(x, ln1w, ln1b, ln1o);

  // 3) fused QKV GEMM with head-split store: [B][H][S][64] bf16
  gemm_bf16_kernel<<<dim3(TOK / 128, (3 * EMB) / 128), 256, LDS_BYTES, stream>>>(
      ln1o, WqkvT, nullptr, nullptr, nullptr, qkv, TOK, 3 * EMB, EMB, GF_QKV);

  // 4) causal flash attention, h = x + ctx
  flash_attn_kernel<<<dim3(SEQ / 16, HEADS, BATCH), 32, 0, stream>>>(
      qkv, qkv + (size_t)TOK * EMB, qkv + (size_t)2 * TOK * EMB, x, hbuf);

  // 5) ln2(h)
  layernorm_bf16_kernel<<<TOK, 256, 0, stream>>>(hbuf, ln2w, ln2b, ln2o);

  // 6) FFN up-projection + GELU -> bf16
  gemm_bf16_kernel<<<dim3(TOK / 128, FF / 128), 256, LDS_BYTES, stream>>>(
      ln2o, w1T, ffb1, nullptr, nullptr, f1, TOK, FF, EMB, GF_GELU | GF_OUTB);

  // 7) FFN down-projection + bias + residual(h) -> bf16 h2
  gemm_bf16_kernel<<<dim3(TOK / 128, EMB / 128), 256, LDS_BYTES, stream>>>(
      f1, w2T, ffb2, hbuf, nullptr, h2b, TOK, EMB, FF, GF_RES | GF_OUTB);

  // 8) vocab projection + bias -> fp32 output
  gemm_bf16_kernel<<<dim3(TOK / 128, VOCAB / 128), 256, LDS_BYTES, stream>>>(
      h2b, projT, projb, nullptr, out, nullptr, TOK, VOCAB, EMB, GF_OUTF);
}
